// DFlashAttentionANE_45853070852106
// MI455X (gfx1250) — matrix operand order repaired and checked by reading, not compiled, not run on hardware
//
#include <hip/hip_runtime.h>
#include <hip/hip_bf16.h>

typedef unsigned short u16;
typedef unsigned int   u32;
typedef unsigned long long u64;
typedef __attribute__((ext_vector_type(16))) __bf16 bf16x16;
typedef __attribute__((ext_vector_type(8)))  float  f32x8;
typedef __attribute__((ext_vector_type(4)))  u32    u32x4;
typedef __attribute__((ext_vector_type(4)))  int    i32x4;
typedef __attribute__((ext_vector_type(8)))  int    i32x8;

#define L_Q   1024
#define S_CTX 4096
#define T_KV  (S_CTX + L_Q)   // 5120
#define HID   4096
#define HQ    32
#define HKV   8
#define REP   (HQ / HKV)
#define HD    128
#define QK_SCALE 0.08838834764831845f   // 128^-0.5
#define RMS_EPS  1e-6f

#if __has_builtin(__builtin_amdgcn_tensor_load_to_lds) && __has_builtin(__builtin_amdgcn_s_wait_tensorcnt)
#define USE_TDM 1
#else
#define USE_TDM 0
#endif

union FragAB { uint4 q[2]; u16 u[16]; bf16x16 v; };

__device__ __forceinline__ u16 f2b(float f) {       // native v_cvt, RNE
  __bf16 b = (__bf16)f;
  return __builtin_bit_cast(u16, b);
}
__device__ __forceinline__ float b2f(u16 h) { return __uint_as_float(((u32)h) << 16); }
__device__ __forceinline__ u32 pack2(u16 lo, u16 hi) { return (u32)lo | ((u32)hi << 16); }

#if USE_TDM
// ---- Tensor Data Mover: 2D tile (row-major, bf16) global -> LDS, optional
// LDS row padding.  Descriptor bitfields per CDNA5 ISA 8.3/8.4.
__device__ __forceinline__ void tdm_load_2d(
    u32 lds_off, const void* gaddr, u32 dim0, u32 dim1,
    u32 tile0, u32 tile1, u64 stride0,
    u32 pad_interval_code, u32 pad_amount_code, u32 pad_en) {
  u64 ga = (u64)(uintptr_t)gaddr;
  u32x4 g0;
  g0[0] = 1u;                                          // count=1 valid descriptor
  g0[1] = lds_off;                                     // lds_addr [63:32]
  g0[2] = (u32)(ga & 0xFFFFFFFFu);                     // global_addr lo
  g0[3] = (u32)((ga >> 32) & 0x1FFFFFFu) | (2u << 30); // global_addr hi | type=2
  i32x8 g1;
  g1[0] = (int)((1u << 16) |                           // data_size=1 (2 bytes)
                (pad_en ? (1u << 20) : 0u) |
                (pad_interval_code << 22) | (pad_amount_code << 25));
  g1[1] = (int)((dim0 & 0xFFFFu) << 16);               // tensor_dim0 lo16 (barrier addr=0)
  g1[2] = (int)(((dim0 >> 16) & 0xFFFFu) | ((dim1 & 0xFFFFu) << 16)); // dim0 hi | dim1 lo
  g1[3] = (int)(((dim1 >> 16) & 0xFFFFu) | ((tile0 & 0xFFFFu) << 16)); // dim1 hi | tile_dim0
  g1[4] = (int)(tile1 & 0xFFFFu);                      // tile_dim1, tile_dim2=0 (2D)
  g1[5] = (int)(u32)(stride0 & 0xFFFFFFFFull);         // tensor_dim0_stride lo32
  g1[6] = (int)(u32)((stride0 >> 32) & 0xFFFFull);     // stride hi16 | dim1_stride lo16=0
  g1[7] = 0;
  i32x4 z4 = {0, 0, 0, 0};
#if __clang_major__ >= 23
  i32x8 z8 = {0, 0, 0, 0, 0, 0, 0, 0};
  __builtin_amdgcn_tensor_load_to_lds(g0, g1, z4, z4, z8, 0);
#else
  __builtin_amdgcn_tensor_load_to_lds(g0, g1, z4, z4, 0);
#endif
}
#endif

// ---------------------------------------------------------------- fp32 -> bf16
// n must be a multiple of 4 (true for every buffer here).
__global__ void cvt_f32_bf16_x4(const float4* __restrict__ in, uint2* __restrict__ out, size_t n4) {
  size_t i = (size_t)blockIdx.x * blockDim.x + threadIdx.x;
  size_t stride = (size_t)gridDim.x * blockDim.x;
  for (; i < n4; i += stride) {
    float4 v = in[i];
    uint2 o;
    o.x = pack2(f2b(v.x), f2b(v.y));
    o.y = pack2(f2b(v.z), f2b(v.w));
    out[i] = o;
  }
}

// ------------------------------------------------------- bf16 WMMA tiled GEMM
// C[M,N] = A[M,K] (row-major bf16) * B[K,N] (row-major bf16), f32 accum.
// 256 threads = 8 waves; block tile 128x128, K-step 32; wave tile 16x128.
// A tile staged by the Tensor Data Mover (padded rows); B tile staged
// transposed by the waves while the TDM runs.
template <bool OUTF32>
__global__ __launch_bounds__(256) void gemm_bf16_wmma(
    const u16* __restrict__ A, const u16* __restrict__ B, void* __restrict__ Cv,
    int M, int N, int K) {
  constexpr int BM = 128, BN = 128, BK = 32, LDA = BK + 8;  // 40-elem stride, 16B aligned
  __shared__ __align__(16) u16 As[BM * LDA];
  __shared__ __align__(16) u16 Bs[BN * LDA];   // stored TRANSPOSED: [n][k]
  const int tid = threadIdx.x, wave = tid >> 5, lane = tid & 31;
  const int m16 = lane & 15, half = lane >> 4;
  const int bm = blockIdx.y * BM, bn = blockIdx.x * BN;
  const int kb = half * 8, kb2 = half * 16;

  f32x8 acc[8] = {};

  for (int k0 = 0; k0 < K; k0 += BK) {
#if USE_TDM
    // One TDM op stages the whole 128x32 A tile; pad_interval=16 DWORDs
    // (one 32-elem row), pad_amount=4 DWORDs (8 elems) -> LDS stride 40.
    if (tid == 0) {
      tdm_load_2d((u32)(uintptr_t)As, A + (size_t)bm * K + k0,
                  (u32)K, (u32)(M - bm), BK, BM, (u64)K,
                  /*pad_interval 16dw*/3, /*pad_amount 4dw*/3, /*pad_en*/1);
    }
#else
    {  // fallback: cooperative A tile load (two b128 per thread)
      int r = tid >> 1, c = (tid & 1) * 16;
      const uint4* src = (const uint4*)(A + (size_t)(bm + r) * K + k0 + c);
      uint4 d0 = src[0], d1 = src[1];
      *(uint4*)&As[r * LDA + c]     = d0;
      *(uint4*)&As[r * LDA + c + 8] = d1;
    }
#endif
    // Stage B tile 32x128, transposed into Bs[n][k] (overlaps with TDM).
    {
      int kk = tid >> 3, ch = (tid & 7) * 16;
      const u16* src = B + (size_t)(k0 + kk) * N + bn + ch;
      u16 tmp[16];
      *(uint4*)tmp       = *(const uint4*)src;
      *(uint4*)(tmp + 8) = *(const uint4*)(src + 8);
#pragma unroll
      for (int j = 0; j < 16; j++) Bs[(ch + j) * LDA + kk] = tmp[j];
    }
    if (k0 + BK < K) {  // gfx1250 global_prefetch_b8 for next K-tile of B
      __builtin_prefetch(B + (size_t)(k0 + BK + (tid >> 3)) * N + bn + (tid & 7) * 16, 0, 0);
    }
#if USE_TDM
    __builtin_amdgcn_s_wait_tensorcnt(0);   // no-op for non-issuing waves
#endif
    __syncthreads();

    // A fragment: lane m = lane&15, K-halves {kb..kb+7, kb+16..kb+23}
    FragAB a;
    const int ar = wave * 16 + m16;
    a.q[0] = *(const uint4*)&As[ar * LDA + kb];
    a.q[1] = *(const uint4*)&As[ar * LDA + kb + 16];
#pragma unroll
    for (int nt = 0; nt < 8; nt++) {
      FragAB b;  // B fragment: lane n, K-halves {kb..kb+7, kb+16..kb+23}
      const int br = nt * 16 + m16;
      b.q[0] = *(const uint4*)&Bs[br * LDA + kb];
      b.q[1] = *(const uint4*)&Bs[br * LDA + kb + 16];
      acc[nt] = __builtin_amdgcn_wmma_f32_16x16x32_bf16(
          false, a.v, false, b.v, (short)0, acc[nt], false, false);
    }
    __syncthreads();
  }

  // Epilogue: D layout — VGPR r holds row r + 8*half, lane holds col n.
#pragma unroll
  for (int nt = 0; nt < 8; nt++) {
#pragma unroll
    for (int r = 0; r < 8; r++) {
      int m = bm + wave * 16 + half * 8 + r;
      int n = bn + nt * 16 + m16;
      float v = acc[nt][r];
      if (OUTF32) ((float*)Cv)[(size_t)m * N + n] = v;
      else        ((u16*)Cv)[(size_t)m * N + n]  = f2b(v);
    }
  }
}

// ----------------------------------------------- per-head RMSNorm + RoPE (bf16)
// proj: [rows][nheads*128] bf16 ; out: [nheads][rows][128] bf16 (head-major)
__global__ __launch_bounds__(128) void normrope_kernel(
    const u16* __restrict__ proj, u16* __restrict__ out,
    const float* __restrict__ cosv, const float* __restrict__ sinv,
    const float* __restrict__ w, int rows, int nheads) {
  __shared__ float sh[HD];
  const int d = threadIdx.x, row = blockIdx.x, h = blockIdx.y;
  float x = b2f(proj[(size_t)row * nheads * HD + h * HD + d]);
  sh[d] = x * x;
  __syncthreads();
  for (int s = 64; s > 0; s >>= 1) { if (d < s) sh[d] += sh[d + s]; __syncthreads(); }
  float rinv = rsqrtf(sh[0] / (float)HD + RMS_EPS);
  __syncthreads();
  float xn = w[d] * x * rinv;
  sh[d] = xn;
  __syncthreads();
  float rot = (d < 64) ? -sh[d + 64] : sh[d - 64];
  float o = xn * cosv[(size_t)row * HD + d] + rot * sinv[(size_t)row * HD + d];
  out[((size_t)h * rows + row) * HD + d] = f2b(o);
}

// --------------------------------------------------- flash attention via WMMA
// Qn: [HQ][L][128] bf16 ; Kn: [HKV][T][128] bf16 ; V: [T][HKV*128] bf16 (raw)
// AO: [L][HQ*128] bf16.  128 threads = 4 waves; wave w owns 16-query tile.
__global__ __launch_bounds__(128) void attn_fa_wmma(
    const u16* __restrict__ Qn, const u16* __restrict__ Kn,
    const u16* __restrict__ V, u16* __restrict__ AO) {
  constexpr int LDV = 40, LDP = 40;
  __shared__ __align__(16) u16 Vs[HD * LDV];       // V^T tile: [d][key] 128x32
  __shared__ __align__(16) u16 Ps[4 * 16 * LDP];   // P tile per wave: [m][key] 16x32
  const int tid = threadIdx.x, wave = tid >> 5, lane = tid & 31;
  const int m16 = lane & 15, half = lane >> 4;
  const int kb = half * 8, kb2 = half * 16;
  const int h = blockIdx.y, hkv = h / REP;
  const int ql = blockIdx.x * 64 + wave * 16 + m16;

  // Q fragments over d=0..127 (4 K-chunks of 32), pre-scaled by 1/sqrt(d)
  FragAB aq[4];
  const u16* qrow = Qn + ((size_t)h * L_Q + ql) * HD;
#pragma unroll
  for (int c = 0; c < 4; c++) {
    u16 t0a[8], t1a[8];
    *(uint4*)t0a = *(const uint4*)&qrow[c * 32 + kb];
    *(uint4*)t1a = *(const uint4*)&qrow[c * 32 + kb + 16];
#pragma unroll
    for (int j = 0; j < 8; j++) {
      aq[c].u[j]     = f2b(b2f(t0a[j]) * QK_SCALE);
      aq[c].u[j + 8] = f2b(b2f(t1a[j]) * QK_SCALE);
    }
  }

  float mrow[8], lrow[8];
  f32x8 acco[8] = {};
#pragma unroll
  for (int r = 0; r < 8; r++) { mrow[r] = -1e30f; lrow[r] = 0.f; }
  u16* Pw = Ps + wave * 16 * LDP;

  for (int t0 = 0; t0 < T_KV; t0 += 32) {
    // Cooperatively stage V^T tile (32 keys x 128 dims -> Vs[d][key])
    {
      int vr = tid >> 2, dblk = (tid & 3) * 32;
      const u16* vrow = V + (size_t)(t0 + vr) * (HKV * HD) + hkv * HD + dblk;
#pragma unroll
      for (int q = 0; q < 4; q++) {
        u16 tmp[8];
        *(uint4*)tmp = *(const uint4*)(vrow + q * 8);
#pragma unroll
        for (int j = 0; j < 8; j++) Vs[(dblk + q * 8 + j) * LDV + vr] = tmp[j];
      }
    }
    __syncthreads();

    // S = Q K^T : two 16x16 key blocks, 4 WMMAs each over d
    f32x8 accs[2] = {};
#pragma unroll
    for (int nt = 0; nt < 2; nt++) {
      const int key = t0 + nt * 16 + m16;
      const u16* krow = Kn + ((size_t)hkv * T_KV + key) * HD;
#pragma unroll
      for (int c = 0; c < 4; c++) {
        FragAB bk;  // B fragment: lane n = key, K = d halves {kb..kb+7, kb+16..kb+23}
        bk.q[0] = *(const uint4*)&krow[c * 32 + kb];
        bk.q[1] = *(const uint4*)&krow[c * 32 + kb + 16];
        accs[nt] = __builtin_amdgcn_wmma_f32_16x16x32_bf16(
            false, aq[c].v, false, bk.v, (short)0, accs[nt], false, false);
      }
    }

    // Online softmax (row = r + 8*half; 16 lanes of a half share a row)
#pragma unroll
    for (int r = 0; r < 8; r++) {
      float s0 = accs[0][r], s1 = accs[1][r];
      float mx = fmaxf(s0, s1);
#pragma unroll
      for (int msk = 8; msk >= 1; msk >>= 1) mx = fmaxf(mx, __shfl_xor(mx, msk, 32));
      float mnew = fmaxf(mrow[r], mx);
      float p0 = __expf(s0 - mnew), p1 = __expf(s1 - mnew);
      float sum = p0 + p1;
#pragma unroll
      for (int msk = 8; msk >= 1; msk >>= 1) sum += __shfl_xor(sum, msk, 32);
      float corr = __expf(mrow[r] - mnew);
      lrow[r] = lrow[r] * corr + sum;
      mrow[r] = mnew;
#pragma unroll
      for (int nt2 = 0; nt2 < 8; nt2++) acco[nt2][r] *= corr;
      const int pm = r + half * 8;
      Pw[pm * LDP + m16]      = f2b(p0);
      Pw[pm * LDP + 16 + m16] = f2b(p1);
    }
    __syncthreads();

    // O += P V : reload P in A-fragment layout, V^T gives contiguous B frags
    FragAB ap;
    ap.q[0] = *(const uint4*)&Pw[m16 * LDP + kb];
    ap.q[1] = *(const uint4*)&Pw[m16 * LDP + kb + 16];
#pragma unroll
    for (int nt2 = 0; nt2 < 8; nt2++) {
      FragAB bv;
      const int d = nt2 * 16 + m16;
      bv.q[0] = *(const uint4*)&Vs[d * LDV + kb];
      bv.q[1] = *(const uint4*)&Vs[d * LDV + kb + 16];
      acco[nt2] = __builtin_amdgcn_wmma_f32_16x16x32_bf16(
          false, ap.v, false, bv.v, (short)0, acco[nt2], false, false);
    }
    __syncthreads();
  }

  // Epilogue: AO[l][h*128+d] = O / l
#pragma unroll
  for (int nt2 = 0; nt2 < 8; nt2++) {
#pragma unroll
    for (int r = 0; r < 8; r++) {
      int li = blockIdx.x * 64 + wave * 16 + half * 8 + r;
      int d  = nt2 * 16 + m16;
      AO[(size_t)li * (HQ * HD) + h * HD + d] = f2b(acco[nt2][r] / lrow[r]);
    }
  }
}

// ------------------------------------------------------------------- launcher
extern "C" void kernel_launch(void* const* d_in, const int* in_sizes, int n_in,
                              void* d_out, int out_size, void* d_ws, size_t ws_size,
                              hipStream_t stream) {
  const float* x     = (const float*)d_in[0];
  const float* x_ctx = (const float*)d_in[1];
  const float* cos_q = (const float*)d_in[2];
  const float* sin_q = (const float*)d_in[3];
  const float* cos_k = (const float*)d_in[4];
  const float* sin_k = (const float*)d_in[5];
  const float* Wq    = (const float*)d_in[6];
  const float* Wk    = (const float*)d_in[7];
  const float* Wv    = (const float*)d_in[8];
  const float* Wo    = (const float*)d_in[9];
  const float* qnw   = (const float*)d_in[10];
  const float* knw   = (const float*)d_in[11];

  // Workspace carve (bf16 buffers; all sizes are multiples of 16B)
  u16* p = (u16*)d_ws;
  u16* xb   = p; p += (size_t)L_Q  * HID;
  u16* cb   = p; p += (size_t)T_KV * HID;
  u16* Wqb  = p; p += (size_t)HID * HQ  * HD;
  u16* Wkb  = p; p += (size_t)HID * HKV * HD;
  u16* Wvb  = p; p += (size_t)HID * HKV * HD;
  u16* Wob  = p; p += (size_t)HQ * HD * HID;
  u16* Qraw = p; p += (size_t)L_Q  * HQ  * HD;
  u16* Kraw = p; p += (size_t)T_KV * HKV * HD;
  u16* Vraw = p; p += (size_t)T_KV * HKV * HD;
  u16* Qn   = p; p += (size_t)HQ  * L_Q  * HD;
  u16* Kn   = p; p += (size_t)HKV * T_KV * HD;
  u16* AO   = p; p += (size_t)L_Q * HQ * HD;

  auto cvt = [&](const float* src, u16* dst, size_t n) {
    size_t n4 = n / 4;
    size_t b = (n4 + 255) / 256;
    int blocks = (b > 16384) ? 16384 : (int)b;
    cvt_f32_bf16_x4<<<blocks, 256, 0, stream>>>((const float4*)src, (uint2*)dst, n4);
  };
  cvt(x,     xb, (size_t)L_Q * HID);
  cvt(x_ctx, cb, (size_t)S_CTX * HID);
  cvt(x,     cb + (size_t)S_CTX * HID, (size_t)L_Q * HID);
  cvt(Wq, Wqb, (size_t)HID * HQ * HD);
  cvt(Wk, Wkb, (size_t)HID * HKV * HD);
  cvt(Wv, Wvb, (size_t)HID * HKV * HD);
  cvt(Wo, Wob, (size_t)HQ * HD * HID);

  // Projections
  gemm_bf16_wmma<false><<<dim3((HQ * HD) / 128, L_Q / 128),  256, 0, stream>>>(xb, Wqb, Qraw, L_Q,  HQ * HD,  HID);
  gemm_bf16_wmma<false><<<dim3((HKV * HD) / 128, T_KV / 128), 256, 0, stream>>>(cb, Wkb, Kraw, T_KV, HKV * HD, HID);
  gemm_bf16_wmma<false><<<dim3((HKV * HD) / 128, T_KV / 128), 256, 0, stream>>>(cb, Wvb, Vraw, T_KV, HKV * HD, HID);

  // RMSNorm + RoPE (head-major repack)
  normrope_kernel<<<dim3(L_Q,  HQ),  128, 0, stream>>>(Qraw, Qn, cos_q, sin_q, qnw, L_Q,  HQ);
  normrope_kernel<<<dim3(T_KV, HKV), 128, 0, stream>>>(Kraw, Kn, cos_k, sin_k, knw, T_KV, HKV);

  // Flash attention
  attn_fa_wmma<<<dim3(L_Q / 64, HQ), 128, 0, stream>>>(Qn, Kn, Vraw, AO);

  // Output projection (fp32 out)
  gemm_bf16_wmma<true><<<dim3(HID / 128, L_Q / 128), 256, 0, stream>>>(AO, Wob, (float*)d_out, L_Q, HID, HQ * HD);
}
